// AdaptiveBasisNet_27032524161616
// MI455X (gfx1250) — matrix operand order repaired and checked by reading, not compiled, not run on hardware
//
#include <hip/hip_runtime.h>
#include <hip/hip_bf16.h>
#include <stdint.h>

// ---------------------------------------------------------------------------
// AdaptiveBasisNet for MI455X (gfx1250, wave32, WMMA).
// Convs = implicit GEMM with v_wmma_f32_16x16x32_f16, f16 activations.
// Weights pre-packed to the LDS image once per layer, then copied into LDS
// with CDNA5 async VMEM->LDS (ASYNCcnt) when the builtins are available.
// ---------------------------------------------------------------------------

#define S      256
#define NVER   53215
#define NAP    29

#ifndef __has_builtin
#define __has_builtin(x) 0
#endif
#if __has_builtin(__builtin_amdgcn_global_load_async_to_lds_b128) && \
    __has_builtin(__builtin_amdgcn_s_wait_asynccnt)
#define USE_ASYNC_LDS 1
#else
#define USE_ASYNC_LDS 0
#endif

typedef __attribute__((ext_vector_type(16))) _Float16 v16h;
typedef __attribute__((ext_vector_type(8)))  float    v8f;
typedef _Float16 h16;

#if USE_ASYNC_LDS
// Builtin signature (from hipcc diagnostic): param1 = AS1 (global) v4i*,
// param2 = AS3 (LDS) v4i*, then imm offset, imm cpol.
typedef int v4i_t __attribute__((vector_size(16)));
typedef __attribute__((address_space(1))) v4i_t* as1_v4i;
typedef __attribute__((address_space(3))) v4i_t* as3_v4i;
#endif

union V16U { uint4 q[2]; v16h h; };

// ---------------- 1) sample per-vertex features from 64x64 feature maps ----
__global__ __launch_bounds__(256) void k_sample_vert(
    const float* __restrict__ feat, const float* __restrict__ vert,
    const float* __restrict__ vis, const int* __restrict__ fmask,
    float* __restrict__ vf)
{
  int idx = blockIdx.x * 256 + threadIdx.x;
  if (idx >= 6 * NVER) return;
  int b = idx / NVER, v = idx - b * NVER;
  const float* vp = vert + ((size_t)b * NVER + v) * 3;
  float vx = vp[0], vy = vp[1], vz = vp[2];
  // gx = (vx+128)/127.5 - 1 ; x = (gx+1)*0.5*63  ==>  x = (vx+128)*63/255
  float x = (vx + 128.f) * (63.f / 255.f);
  float y = (128.f - vy) * (63.f / 255.f);
  float x0f = floorf(x), y0f = floorf(y);
  float wx = x - x0f, wy = y - y0f;
  int x0 = (int)x0f, y0 = (int)y0f;
  bool ox0 = (x0 >= 0) && (x0 <= 63), ox1 = (x0 + 1 >= 0) && (x0 + 1 <= 63);
  bool oy0 = (y0 >= 0) && (y0 <= 63), oy1 = (y0 + 1 >= 0) && (y0 + 1 <= 63);
  int xc0 = min(max(x0, 0), 63), xc1 = min(max(x0 + 1, 0), 63);
  int yc0 = min(max(y0, 0), 63), yc1 = min(max(y0 + 1, 0), 63);
  float w00 = (1.f - wx) * (1.f - wy) * ((ox0 && oy0) ? 1.f : 0.f);
  float w10 = wx * (1.f - wy)        * ((ox1 && oy0) ? 1.f : 0.f);
  float w01 = (1.f - wx) * wy        * ((ox0 && oy1) ? 1.f : 0.f);
  float w11 = wx * wy                * ((ox1 && oy1) ? 1.f : 0.f);
  int o00 = yc0 * 64 + xc0, o10 = yc0 * 64 + xc1;
  int o01 = yc1 * 64 + xc0, o11 = yc1 * 64 + xc1;
  // vfm = vert_feat*vis + vert_feat*nonface = vert_feat*(vis + !face)
  float fac = vis[(size_t)b * NVER + v] + (fmask[v] ? 0.f : 1.f);
  const float* fb = feat + (size_t)b * 64 * 4096;
  #pragma unroll 4
  for (int c = 0; c < 64; ++c) {
    const float* f = fb + c * 4096;
    float s = f[o00] * w00 + f[o10] * w10 + f[o01] * w01 + f[o11] * w11;
    vf[((size_t)b * 67 + c) * NVER + v] = s * fac;
  }
  // vert_pos = vert / (H+W) * 2 = vert / 256
  vf[((size_t)b * 67 + 64) * NVER + v] = vx * (1.f / 256.f);
  vf[((size_t)b * 67 + 65) * NVER + v] = vy * (1.f / 256.f);
  vf[((size_t)b * 67 + 66) * NVER + v] = vz * (1.f / 256.f);
}

// ---------------- 2) per-pixel 3-vertex weighted gather -> UV map (f16) ----
__global__ __launch_bounds__(256) void k_build_uvf(
    const float* __restrict__ vf, const int* __restrict__ pidx,
    const float* __restrict__ pw, h16* __restrict__ uvf)
{
  int i = blockIdx.x * 256 + threadIdx.x;     // 6*65536 exact
  int b = i >> 16, p = i & 65535;
  int i0 = pidx[p * 3 + 0], i1 = pidx[p * 3 + 1], i2 = pidx[p * 3 + 2];
  float w0 = pw[p * 3 + 0], w1 = pw[p * 3 + 1], w2 = pw[p * 3 + 2];
  const float* vb = vf + (size_t)b * 67 * NVER;
  for (int c = 0; c < 67; ++c) {
    const float* r = vb + (size_t)c * NVER;
    float s = r[i0] * w0 + r[i1] * w1 + r[i2] * w2;
    uvf[((size_t)(b * 96 + c) << 16) + p] = (h16)s;
  }
  for (int c = 67; c < 96; ++c)               // zero-pad to 96 ch (K mult of 32)
    uvf[((size_t)(b * 96 + c) << 16) + p] = (h16)0.f;
}

// ---------------- 3a) pre-pack conv weights to padded f16 LDS image --------
// wpad layout = [64 c_out][ASTR = KS*KS*CINP + 8] f16 (K ordering: tap-major,
// channel minor, zero pad for ci >= cin_real and the +8 tail).
__global__ void k_prep_w(const float* __restrict__ w, h16* __restrict__ wpad,
                         int cin_real, int KS, int CINP)
{
  int KT = KS * KS * CINP, ASTR = KT + 8;
  int i = blockIdx.x * 256 + threadIdx.x;
  if (i >= 64 * ASTR) return;
  int m = i / ASTR, k = i - m * ASTR;
  float val = 0.f;
  if (k < KT) {
    int tap = k / CINP, ci = k - tap * CINP;
    if (ci < cin_real) {
      int ky = tap / KS, kx = tap - ky * KS;
      val = w[((m * cin_real + ci) * KS + ky) * KS + kx];
    }
  }
  wpad[i] = (h16)val;
}

// ---------------- 3b) implicit-GEMM conv via WMMA f16 ----------------------
// Block = 256 thr = 8 waves: 4 M-tiles (c_out) x 2 N-tiles (pixels).
// Block output tile: 64 c_out x 64 pixels x ROWS rows (amortizes weight
// staging; one A-fragment feeds 2*ROWS WMMAs).
template<int KS, int CINP, int ROWS, int DIL>
__global__ __launch_bounds__(256) void k_conv_wmma(
    const h16* __restrict__ in, const h16* __restrict__ wpad,
    h16* __restrict__ out)
{
  constexpr int KT   = KS * KS * CINP;
  constexpr int ASTR = KT + 8;                     // +8h pad vs bank conflicts
  constexpr int W64  = (KS == 3) ? (64 + 2 * DIL) : 64;
  constexpr int BSTR = CINP + 8;
  constexpr int BPI  = (256 / ROWS) * 4;           // blocks per image
  __shared__ alignas(16) h16 lA[64 * ASTR];        // weights  [c_out][K]
  __shared__ alignas(16) h16 lB[ROWS * KS * W64 * BSTR];  // input [row][col][cin]

  const int tid = threadIdx.x;
  const int bx  = blockIdx.x;
  const int b   = bx / BPI;
  const int rem = bx - b * BPI;
  const int y0  = (rem >> 2) * ROWS;
  const int x0  = (rem & 3) << 6;

  // ---- stage weights: contiguous copy of the prepped f16 image ----
  constexpr int NQ = 64 * ASTR / 8;                // 16B chunks
  const uint4* wq = (const uint4*)wpad;
#if USE_ASYNC_LDS
  for (int e = tid; e < NQ; e += 256)
    __builtin_amdgcn_global_load_async_to_lds_b128(
        (as1_v4i)(wq + e), (as3_v4i)(lA + e * 8), 0, 0);
#else
  for (int e = tid; e < NQ; e += 256)
    *(uint4*)(lA + e * 8) = wq[e];
#endif

  // ---- stage input patch (zero pad at borders), channel-contiguous ----
  constexpr int TOT = ROWS * KS * CINP * W64;
  for (int e = tid; e < TOT; e += 256) {
    int rr = e / (CINP * W64);                     // 0..ROWS*KS-1
    int q  = e - rr * (CINP * W64);
    int ci = q / W64;
    int c  = q - ci * W64;
    int i  = rr / KS, r = rr - i * KS;
    int yy = y0 + i + ((KS == 3) ? DIL * (r - 1) : 0);
    int xx = x0 + c - ((KS == 3) ? DIL : 0);
    h16 val = (h16)0.f;
    if (yy >= 0 && yy < S && xx >= 0 && xx < S)
      val = in[((size_t)(b * CINP + ci) << 16) + (yy << 8) + xx];
    lB[(rr * W64 + c) * BSTR + ci] = val;
  }
#if USE_ASYNC_LDS
  __builtin_amdgcn_s_wait_asynccnt(0);             // weights landed in LDS
#endif
  __syncthreads();

  const int wave = tid >> 5, lane = tid & 31;
  const int wm = wave >> 1, wn = wave & 1;
  const int m0 = wm << 4;
  const int lh = lane >> 4, lm = lane & 15;

  const v8f vzero = {0, 0, 0, 0, 0, 0, 0, 0};
  v8f acc[ROWS][2];
  #pragma unroll
  for (int i = 0; i < ROWS; ++i) { acc[i][0] = vzero; acc[i][1] = vzero; }

  // A frag (16x32 f16): lane<16 -> K 0..15, lane>=16 -> K 16..31
  const h16* Arow = lA + (m0 + lm) * ASTR + lh * 16;
  const int  bK   = lh * 16;  // B frag: lane<16 -> K 0..15, lane>=16 -> 16..31

  #pragma unroll
  for (int tap = 0; tap < KS * KS; ++tap) {
    const int ky = tap / KS, kx = tap - ky * KS;
    const h16* Acol = Arow + tap * CINP;
    #pragma unroll
    for (int kc = 0; kc < CINP; kc += 32) {
      V16U a;
      a.q[0] = *(const uint4*)(Acol + kc);
      a.q[1] = *(const uint4*)(Acol + kc + 8);
      #pragma unroll
      for (int i = 0; i < ROWS; ++i) {
        const int colb = (i * KS + ky) * W64 + wn * 32 + lm +
                         ((KS == 3) ? kx * DIL : 0);
        const h16* B0 = lB + colb * BSTR + bK + kc;
        V16U b0, b1;
        b0.q[0] = *(const uint4*)(B0);
        b0.q[1] = *(const uint4*)(B0 + 8);
        const h16* B1 = B0 + 16 * BSTR;
        b1.q[0] = *(const uint4*)(B1);
        b1.q[1] = *(const uint4*)(B1 + 8);
        acc[i][0] = __builtin_amdgcn_wmma_f32_16x16x32_f16(
            false, a.h, false, b0.h, (short)0, acc[i][0], false, false);
        acc[i][1] = __builtin_amdgcn_wmma_f32_16x16x32_f16(
            false, a.h, false, b1.h, (short)0, acc[i][1], false, false);
      }
    }
  }

  // C/D layout: lane<16 -> M=m0+r, lane>=16 -> M=m0+8+r ; N = x0+wn*32+lm(+16)
  const int co = m0 + lh * 8;
  const int xo = x0 + wn * 32 + lm;
  #pragma unroll
  for (int i = 0; i < ROWS; ++i) {
    size_t ob = (((size_t)b * 64 + co) << 16) + (size_t)((y0 + i) << 8);
    #pragma unroll
    for (int r = 0; r < 8; ++r) {
      out[ob + ((size_t)r << 16) + xo]      = (h16)acc[i][0][r];
      out[ob + ((size_t)r << 16) + xo + 16] = (h16)acc[i][1][r];
    }
  }
}

// ---------------- 4) GroupNorm: zero / reduce / fused apply ----------------
__global__ void k_zero(float* p, int n)
{
  int i = blockIdx.x * blockDim.x + threadIdx.x;
  if (i < n) p[i] = 0.f;
}

__global__ __launch_bounds__(256) void k_gn_reduce(
    const h16* __restrict__ x, float* __restrict__ sums)
{
  int bc = blockIdx.x;                 // b*64 + c
  int b = bc >> 6, c = bc & 63, g = c >> 3;
  const h16* p = x + ((size_t)bc << 16);
  float s = 0.f, ss = 0.f;
  for (int i = threadIdx.x; i < 65536; i += 256) {
    float v = (float)p[i]; s += v; ss += v * v;
  }
  __shared__ float rs[256], rq[256];
  rs[threadIdx.x] = s; rq[threadIdx.x] = ss;
  __syncthreads();
  for (int o = 128; o > 0; o >>= 1) {
    if (threadIdx.x < o) { rs[threadIdx.x] += rs[threadIdx.x + o];
                           rq[threadIdx.x] += rq[threadIdx.x + o]; }
    __syncthreads();
  }
  if (threadIdx.x == 0) {
    atomicAdd(&sums[(b * 8 + g) * 2 + 0], rs[0]);
    atomicAdd(&sums[(b * 8 + g) * 2 + 1], rq[0]);
  }
}

__global__ __launch_bounds__(256) void k_gn_apply(
    const h16* __restrict__ x, const float* __restrict__ sums,
    const float* __restrict__ gamma, const float* __restrict__ beta,
    const h16* __restrict__ res, h16* __restrict__ out, int relu)
{
  int i = blockIdx.x * 256 + threadIdx.x;
  int c = (i >> 16) & 63;
  int b = i >> 22;
  int bg = b * 8 + (c >> 3);
  const float M = 524288.f;            // 8 ch * 65536 px per group
  float mean = sums[bg * 2] / M;
  float var  = sums[bg * 2 + 1] / M - mean * mean;
  float rstd = rsqrtf(var + 1e-5f);
  float v = ((float)x[i] - mean) * rstd * gamma[c] + beta[c];
  if (res)  v += (float)res[i];
  if (relu) v = fmaxf(v, 0.f);
  out[i] = (h16)v;
}

// ---------------- 5) max over 3 views --------------------------------------
__global__ __launch_bounds__(256) void k_maxview(
    const h16* __restrict__ h, h16* __restrict__ o)
{
  int i = blockIdx.x * 256 + threadIdx.x;       // < 2*64*65536 exact
  int n = i >> 22, r = i & ((1 << 22) - 1);
  size_t base = ((size_t)(n * 3)) << 22;
  float a  = (float)h[base + r];
  float b1 = (float)h[base + (1u << 22) + r];
  float c  = (float)h[base + (2u << 22) + r];
  o[i] = (h16)fmaxf(a, fmaxf(b1, c));
}

// ---------------- 6) final 1x1 conv 64 -> 87 (f32 out = adap_B_uv) ---------
__global__ __launch_bounds__(256) void k_conv1x1_87(
    const h16* __restrict__ x, const float* __restrict__ w,
    float* __restrict__ out)
{
  __shared__ h16 lw[87 * 64];
  for (int e = threadIdx.x; e < 87 * 64; e += 256) lw[e] = (h16)w[e];
  __syncthreads();
  int b = blockIdx.y;
  int p = blockIdx.x * 256 + threadIdx.x;
  float xi[64];
  #pragma unroll
  for (int ci = 0; ci < 64; ++ci)
    xi[ci] = (float)x[(((size_t)b * 64 + ci) << 16) + p];
  for (int co = 0; co < 87; ++co) {
    float s = 0.f;
    #pragma unroll
    for (int ci = 0; ci < 64; ++ci) s += xi[ci] * (float)lw[co * 64 + ci];
    out[(((size_t)b * 87 + co) << 16) + p] = s;
  }
}

// ---------------- 7) sample adap_B_uv at uv_coords -> adap_B ---------------
__global__ __launch_bounds__(256) void k_sample_adapB(
    const float* __restrict__ abuv, const float* __restrict__ uvc,
    float* __restrict__ outB)
{
  int idx = blockIdx.x * 256 + threadIdx.x;
  if (idx >= 2 * NVER) return;
  int n = idx / NVER, v = idx - n * NVER;
  // gu = uv/127.5-1 ; x = (gu+1)*0.5*255 = uv  (exact identity)
  float x = uvc[v * 3 + 0], y = uvc[v * 3 + 1];
  float x0f = floorf(x), y0f = floorf(y);
  float wx = x - x0f, wy = y - y0f;
  int x0 = (int)x0f, y0 = (int)y0f;
  bool ox0 = (x0 >= 0) && (x0 <= 255), ox1 = (x0 + 1 >= 0) && (x0 + 1 <= 255);
  bool oy0 = (y0 >= 0) && (y0 <= 255), oy1 = (y0 + 1 >= 0) && (y0 + 1 <= 255);
  int xc0 = min(max(x0, 0), 255), xc1 = min(max(x0 + 1, 0), 255);
  int yc0 = min(max(y0, 0), 255), yc1 = min(max(y0 + 1, 0), 255);
  float w00 = (1.f - wx) * (1.f - wy) * ((ox0 && oy0) ? 1.f : 0.f);
  float w10 = wx * (1.f - wy)        * ((ox1 && oy0) ? 1.f : 0.f);
  float w01 = (1.f - wx) * wy        * ((ox0 && oy1) ? 1.f : 0.f);
  float w11 = wx * wy                * ((ox1 && oy1) ? 1.f : 0.f);
  int o00 = yc0 * 256 + xc0, o10 = yc0 * 256 + xc1;
  int o01 = yc1 * 256 + xc0, o11 = yc1 * 256 + xc1;
  float* ob = outB + ((size_t)n * NVER + v) * 87;       // (n, NVER*3, 29) flat
  const float* ab = abuv + (size_t)n * 87 * 65536;
  for (int c = 0; c < 87; ++c) {
    const float* f = ab + ((size_t)c << 16);
    ob[c] = f[o00] * w00 + f[o10] * w10 + f[o01] * w01 + f[o11] * w11;
  }
}

// ---------------------------------------------------------------------------
extern "C" void kernel_launch(void* const* d_in, const int* in_sizes, int n_in,
                              void* d_out, int out_size, void* d_ws, size_t ws_size,
                              hipStream_t stream)
{
  (void)in_sizes; (void)n_in; (void)out_size; (void)ws_size;
  // d_in layout: setup_inputs() insertion order; params dict leaves assumed
  // flattened in insertion order.
  enum {
    IN_FEAT = 0, IN_VERT, IN_VIS, IN_HIMG, IN_WIMG,
    P_B1_CONV1, P_B1_GN1_W, P_B1_GN1_B, P_B1_CONV2, P_B1_GN2_W, P_B1_GN2_B,
    P_B2_CONV1, P_B2_GN1_W, P_B2_GN1_B, P_B2_CONV2, P_B2_GN2_W, P_B2_GN2_B,
    P_B3_CONV1, P_B3_GN1_W, P_B3_GN1_B, P_B3_CONV2, P_B3_GN2_W, P_B3_GN2_B,
    P_B4_CONV1, P_B4_GN1_W, P_B4_GN1_B, P_B4_CONV2, P_B4_GN2_W, P_B4_GN2_B,
    P_SKIP_CONV, P_SKIP_GN_W, P_SKIP_GN_B,
    P_NET_CONV1, P_NET_GN1_W, P_NET_GN1_B,
    P_NET_CONV2, P_NET_GN2_W, P_NET_GN2_B,
    P_NET_CONV3,
    IN_PIDX, IN_PW, IN_UVC, IN_FMASK
  };
  auto F = [&](int i) { return (const float*)d_in[i]; };

  // ---- workspace carve (~313 MB) ----
  char* ws = (char*)d_ws;
  size_t off = 0;
  auto carve = [&](size_t bytes) -> char* {
    char* p = ws + off; off += (bytes + 255) & ~(size_t)255; return p;
  };
  float* vf   = (float*)carve((size_t)6 * 67 * NVER * 4);     // 85.6 MB
  h16*  uvf   = (h16*) carve((size_t)6 * 96 * 65536 * 2);     // 75.5 MB
  h16*  bufX  = (h16*) carve((size_t)6 * 64 * 65536 * 2);     // 50.3 MB
  h16*  bufT  = (h16*) carve((size_t)6 * 64 * 65536 * 2);
  h16*  bufR  = (h16*) carve((size_t)6 * 64 * 65536 * 2);
  float* stats = (float*)carve(512);
  h16*  wpad  = (h16*) carve((size_t)64 * (9 * 96 + 8) * 2);  // 112 KB max
  h16*  bufU  = (h16*)vf;   // alias: vf dead after k_build_uvf

  float* adapB = (float*)d_out;                 // (2, NVER*3, 29)
  float* abuv  = adapB + (size_t)2 * NVER * 87; // (2, 87, 256, 256)

  // 1) vertex feature sampling (+vis/nonface scaling, +vert_pos channels)
  k_sample_vert<<<(6 * NVER + 255) / 256, 256, 0, stream>>>(
      F(IN_FEAT), F(IN_VERT), F(IN_VIS), (const int*)d_in[IN_FMASK], vf);
  // 2) UV gather into 96-ch (padded) f16 map
  k_build_uvf<<<6 * 65536 / 256, 256, 0, stream>>>(
      vf, (const int*)d_in[IN_PIDX], F(IN_PW), uvf);

  auto gn = [&](const h16* x, int gw, int gb, const h16* res, h16* o,
                int relu, int nv) {
    k_zero<<<1, 128, 0, stream>>>(stats, nv * 16);
    k_gn_reduce<<<nv * 64, 256, 0, stream>>>(x, stats);
    k_gn_apply<<<nv * 64 * 65536 / 256, 256, 0, stream>>>(
        x, stats, F(gw), F(gb), res, o, relu);
  };
  auto prep = [&](int widx, int cin_real, int KS, int CINP) {
    int total = 64 * (KS * KS * CINP + 8);
    k_prep_w<<<(total + 255) / 256, 256, 0, stream>>>(
        F(widx), wpad, cin_real, KS, CINP);
  };
  // 3x3, 64-ch conv with runtime-dispatched compile-time dilation
  auto conv64 = [&](const h16* in, int widx, h16* o, int dil, int nv) {
    prep(widx, 64, 3, 64);
    int grid = nv * 512;                        // ROWS=2 -> 128*4 blocks/img
    if (dil == 1)      k_conv_wmma<3, 64, 2, 1><<<grid, 256, 0, stream>>>(in, wpad, o);
    else if (dil == 2) k_conv_wmma<3, 64, 2, 2><<<grid, 256, 0, stream>>>(in, wpad, o);
    else if (dil == 4) k_conv_wmma<3, 64, 2, 4><<<grid, 256, 0, stream>>>(in, wpad, o);
    else               k_conv_wmma<3, 64, 2, 8><<<grid, 256, 0, stream>>>(in, wpad, o);
  };

  // ---- block 1 (dil 1, skip path) ----
  prep(P_B1_CONV1, 67, 3, 96);
  k_conv_wmma<3, 96, 1, 1><<<6 * 1024, 256, 0, stream>>>(uvf, wpad, bufU);
  gn(bufU, P_B1_GN1_W, P_B1_GN1_B, nullptr, bufT, 1, 6);
  conv64(bufT, P_B1_CONV2, bufU, 1, 6);
  prep(P_SKIP_CONV, 67, 1, 96);
  k_conv_wmma<1, 96, 2, 1><<<6 * 512, 256, 0, stream>>>(uvf, wpad, bufT);
  gn(bufT, P_SKIP_GN_W, P_SKIP_GN_B, nullptr, bufR, 0, 6);
  gn(bufU, P_B1_GN2_W, P_B1_GN2_B, bufR, bufX, 1, 6);          // h = bufX

  // ---- block 2 (dil 2) ----
  conv64(bufX, P_B2_CONV1, bufU, 2, 6);
  gn(bufU, P_B2_GN1_W, P_B2_GN1_B, nullptr, bufT, 1, 6);
  conv64(bufT, P_B2_CONV2, bufU, 2, 6);
  gn(bufU, P_B2_GN2_W, P_B2_GN2_B, bufX, bufR, 1, 6);          // h = bufR

  // ---- block 3 (dil 4) ----
  conv64(bufR, P_B3_CONV1, bufU, 4, 6);
  gn(bufU, P_B3_GN1_W, P_B3_GN1_B, nullptr, bufT, 1, 6);
  conv64(bufT, P_B3_CONV2, bufU, 4, 6);
  gn(bufU, P_B3_GN2_W, P_B3_GN2_B, bufR, bufX, 1, 6);          // h = bufX

  // ---- block 4 (dil 8) ----
  conv64(bufX, P_B4_CONV1, bufU, 8, 6);
  gn(bufU, P_B4_GN1_W, P_B4_GN1_B, nullptr, bufT, 1, 6);
  conv64(bufT, P_B4_CONV2, bufU, 8, 6);
  gn(bufU, P_B4_GN2_W, P_B4_GN2_B, bufX, bufR, 1, 6);          // h = bufR

  // ---- max over 3 views -> (2,64,256,256) in bufX ----
  k_maxview<<<2 * 64 * 65536 / 256, 256, 0, stream>>>(bufR, bufX);

  // ---- refinement net (nv = 2) ----
  conv64(bufX, P_NET_CONV1, bufU, 1, 2);
  gn(bufU, P_NET_GN1_W, P_NET_GN1_B, nullptr, bufT, 1, 2);
  conv64(bufT, P_NET_CONV2, bufU, 1, 2);
  gn(bufU, P_NET_GN2_W, P_NET_GN2_B, nullptr, bufX, 1, 2);
  k_conv1x1_87<<<dim3(256, 2), 256, 0, stream>>>(bufX, F(P_NET_CONV3), abuv);

  // ---- final vertex sampling of adap_B_uv ----
  k_sample_adapB<<<(2 * NVER + 255) / 256, 256, 0, stream>>>(abuv, F(IN_UVC), adapB);
}
